// PhysicalSpecialist3D_41669772706317
// MI455X (gfx1250) — matrix operand order repaired and checked by reading, not compiled, not run on hardware
//
#include <hip/hip_runtime.h>
#include <hip/hip_bf16.h>

typedef _Float16 f16;
typedef __attribute__((ext_vector_type(16))) _Float16 v16h;
typedef __attribute__((ext_vector_type(8)))  _Float16 v8h;
typedef __attribute__((ext_vector_type(8)))  float    v8f;
typedef __attribute__((ext_vector_type(4)))  int      v4i;
typedef __attribute__((ext_vector_type(8)))  int      v8i;
typedef __attribute__((ext_vector_type(4)))  unsigned int v4u;

#define NN 4096
#define HH 128
#define KM 32
#define KI 20

static inline int cdiv(int a, int b) { return (a + b - 1) / b; }

// ---------------- async global->LDS staging (CDNA5, ASYNCcnt) ----------------

#if defined(__HIP_DEVICE_COMPILE__) && __has_builtin(__builtin_amdgcn_global_load_async_to_lds_b128)
#define HAVE_ASYNC_LDS 1
#else
#define HAVE_ASYNC_LDS 0
#endif

__device__ __forceinline__ void stage_bulk(const f16* __restrict__ g, f16* s, int halves) {
#if HAVE_ASYNC_LDS
  // one b128 (8 halves) per lane per iteration, ASYNCcnt-tracked DMA into LDS
  for (int i = threadIdx.x * 8; i < halves; i += 256 * 8) {
    void* gg = (void*)(g + i);
    void* ss = (void*)(s + i);
    __builtin_amdgcn_global_load_async_to_lds_b128(
        (__attribute__((address_space(1))) v4i*)gg,
        (__attribute__((address_space(3))) v4i*)ss, 0, 0);
  }
#else
  const uint4* gv = (const uint4*)g;
  uint4* sv = (uint4*)s;
  for (int i = threadIdx.x; i < (halves >> 3); i += 256) sv[i] = gv[i];
#endif
}

__device__ __forceinline__ void stage_wait_barrier() {
#if HAVE_ASYNC_LDS
#if __has_builtin(__builtin_amdgcn_s_wait_asynccnt)
  __builtin_amdgcn_s_wait_asynccnt(0);
#else
  asm volatile("s_wait_asynccnt 0" ::: "memory");
#endif
#endif
  __syncthreads();
}

// ---------------- Tensor Data Mover staging (CDNA5, TENSORcnt) ----------------

#if defined(__HIP_DEVICE_COMPILE__) && __has_builtin(__builtin_amdgcn_tensor_load_to_lds)
#define HAVE_TDM 1
#else
#define HAVE_TDM 0
#endif

#if HAVE_TDM
__device__ __forceinline__ unsigned lds_offset_of(void* p) {
  return (unsigned)(unsigned long long)(__attribute__((address_space(3))) void*)p;
}

// 1-D contiguous tile load via TDM: D# group0/group1 per ISA 8.3/8.4,
// data_size=8B, tensor_dim0 = tile_dim0 = bytes/8, groups 2/3 unused (<=2D).
__device__ __forceinline__ void tdm_load(const void* gsrc, void* ldst, int bytes) {
  unsigned long long ga = (unsigned long long)(size_t)gsrc;
  unsigned lo = lds_offset_of(ldst);
  int units = bytes >> 3;
  v4u g0;
  g0[0] = 1u;                                   // count=1 (valid user descriptor)
  g0[1] = lo;                                   // lds_addr
  g0[2] = (unsigned)ga;                         // global_addr[31:0]
  g0[3] = (unsigned)((ga >> 32) & 0x1ffffffull) | (2u << 30);  // addr[56:32] | type=2
  v8i g1;
  g1[0] = 3 << 16;                              // wg_mask=0, data_size=3 (8B)
  g1[1] = (units & 0xffff) << 16;               // tensor_dim0[15:0] at bits 63:48
  g1[2] = ((units >> 16) & 0xffff) | (1 << 16); // tensor_dim0[31:16], tensor_dim1=1
  g1[3] = (units & 0xffff) << 16;               // tile_dim0 at bits 127:112
  g1[4] = 0;                                    // tile_dim1/2 unused
  g1[5] = units;                                // tensor_dim0_stride[31:0]
  g1[6] = 0;
  g1[7] = 0;
  v4i z4 = {0, 0, 0, 0};
#if __clang_major__ >= 23
  v8i z8 = {0, 0, 0, 0, 0, 0, 0, 0};
  __builtin_amdgcn_tensor_load_to_lds(g0, g1, z4, z4, z8, 0);
#else
  __builtin_amdgcn_tensor_load_to_lds(g0, g1, z4, z4, 0);
#endif
}
#endif  // HAVE_TDM

// stage a pair of 128x128 f16 weight buffers into LDS (TDM if available)
__device__ __forceinline__ void stage_wpair(const f16* gA, f16* sA_, const f16* gB, f16* sB_) {
#if HAVE_TDM
  if ((threadIdx.x >> 5) == 0) {   // one wave issues the DMAs (TDM ignores EXEC)
    tdm_load(gA, sA_, 128 * 128 * 2);
    tdm_load(gB, sB_, 128 * 128 * 2);
  }
#else
  stage_bulk(gA, sA_, 128 * 128);
  stage_bulk(gB, sB_, 128 * 128);
#endif
}

__device__ __forceinline__ void stage_wpair_wait() {
#if HAVE_TDM
  if ((threadIdx.x >> 5) == 0) {
#if __has_builtin(__builtin_amdgcn_s_wait_tensorcnt)
    __builtin_amdgcn_s_wait_tensorcnt(0);
#else
    asm volatile("s_wait_tensorcnt 0" ::: "memory");
#endif
  }
  __syncthreads();
#else
  stage_wait_barrier();
#endif
}

// ---------------- device helpers ----------------

__device__ __forceinline__ float siluf(float x) { return x / (1.f + __expf(-x)); }

__device__ __forceinline__ v8f zero8() {
  v8f z;
#pragma unroll
  for (int q = 0; q < 8; ++q) z[q] = 0.f;
  return z;
}

__device__ __forceinline__ v8f wmma16(v16h a, v16h b, v8f c) {
  // D = A(16x32 f16) * B(32x16 f16) + C(16x16 f32)
  return __builtin_amdgcn_wmma_f32_16x16x32_f16(false, a, false, b, (short)0, c, false, false);
}

// A fragment: 16x32 f16 tile at A (row-major, leading dim lda).
__device__ __forceinline__ v16h load_a_frag(const f16* A, int lda) {
  int lane = threadIdx.x & 31;
  const f16* p = A + (lane & 15) * lda + ((lane >> 4) << 4);
  v8h lo = *(const v8h*)p;
  v8h hi = *(const v8h*)(p + 8);
  return __builtin_shufflevector(lo, hi, 0, 1, 2, 3, 4, 5, 6, 7, 8, 9, 10, 11, 12, 13, 14, 15);
}

// B fragment from pre-swizzled weights: fragment `frag` holds a 32x16 tile laid
// out as [lane][16 contiguous halves] -> two ds_load_b128 per fragment.
__device__ __forceinline__ v16h load_b_swz(const f16* Wswz, int frag) {
  int lane = threadIdx.x & 31;
  const f16* p = Wswz + ((size_t)(frag * 32 + lane) << 4);
  v8h lo = *(const v8h*)p;
  v8h hi = *(const v8h*)(p + 8);
  return __builtin_shufflevector(lo, hi, 0, 1, 2, 3, 4, 5, 6, 7, 8, 9, 10, 11, 12, 13, 14, 15);
}

// ---------------- converts ----------------

__global__ void cvt_kernel(const float* __restrict__ s, f16* __restrict__ d, int n) {
  int i = blockIdx.x * blockDim.x + threadIdx.x;
  if (i < n) d[i] = (f16)s[i];
}

// f32 [K][Nn] row-major -> f16 swizzled into per-fragment WMMA-B register layout.
__global__ void cvt_swz_kernel(const float* __restrict__ s, f16* __restrict__ d, int K, int Nn) {
  int id = blockIdx.x * blockDim.x + threadIdx.x;
  if (id >= K * Nn) return;
  int k = id / Nn, n = id - k * Nn;
  int kt = k >> 5, kk = k & 31, nt = n >> 4, nn = n & 15;
  int lane = (kk & 16) | nn;
  int e = kk & 15;
  int frag = kt * (Nn >> 4) + nt;
  d[(((frag << 5) + lane) << 4) + e] = (f16)s[id];
}

// ---------------- neighbor search ----------------

#define MAXC 1024

__global__ __launch_bounds__(128) void neighbors_kernel(
    const float* __restrict__ pos, const int* __restrict__ batch,
    float cutoff2, int K, int* __restrict__ idx, float* __restrict__ maskf) {
  __shared__ float sD[4][MAXC];
  __shared__ int   sI[4][MAXC];
  __shared__ int   sN[4];
  int wave = threadIdx.x >> 5, lane = threadIdx.x & 31;
  int i = blockIdx.x * 4 + wave;
  if (i >= NN) return;
  int b = batch[i];
  int l = 0, r = NN;
  while (l < r) { int m = (l + r) >> 1; if (batch[m] < b) l = m + 1; else r = m; }
  int lo = l;
  l = lo; r = NN;
  while (l < r) { int m = (l + r) >> 1; if (batch[m] <= b) l = m + 1; else r = m; }
  int hi = l;

  if (lane == 0) sN[wave] = 0;
  asm volatile("s_wait_dscnt 0" ::: "memory");

  float px = pos[i * 3], py = pos[i * 3 + 1], pz = pos[i * 3 + 2];
  for (int c = lo + lane; c < hi; c += 32) {
    if (c == i) continue;
    float dx = px - pos[c * 3], dy = py - pos[c * 3 + 1], dz = pz - pos[c * 3 + 2];
    float d2 = dx * dx + dy * dy + dz * dz;
    if (d2 <= cutoff2) {
      int s = atomicAdd(&sN[wave], 1);
      if (s < MAXC) { sD[wave][s] = d2; sI[wave][s] = c; }
    }
  }
  asm volatile("s_wait_dscnt 0" ::: "memory");
  int nc = sN[wave]; if (nc > MAXC) nc = MAXC;

  for (int k = 0; k < K; ++k) {
    float best = 3.0e38f; int bi = -1;
    for (int c = lane; c < nc; c += 32) {
      float d = sD[wave][c];
      if (d < best) { best = d; bi = c; }
    }
#pragma unroll
    for (int off = 16; off > 0; off >>= 1) {
      float ob = __shfl_xor(best, off, 32);
      int   oi = __shfl_xor(bi, off, 32);
      if (oi >= 0 && (ob < best || (ob == best && (bi < 0 || oi < bi)))) { best = ob; bi = oi; }
    }
    bool valid = (bi >= 0);
    if (lane == 0) {
      idx[i * KM + k]   = valid ? sI[wave][bi] : i;
      maskf[i * KM + k] = valid ? 1.f : 0.f;
      if (valid) sD[wave][bi] = 3.0e38f;
    }
    asm volatile("s_wait_dscnt 0" ::: "memory");
  }
}

// ---------------- generic node GEMM: C = act(A[M,128] @ W[128,Nn] + bias) ----------------

template <int ACT>
__global__ __launch_bounds__(256) void gemm_node_kernel(
    const f16* __restrict__ A, const f16* __restrict__ Wswz, const float* __restrict__ bias,
    float* __restrict__ C32, f16* __restrict__ C16, int ldc, int M, int Nn) {
  __shared__ f16 sW[128 * 128];
  stage_bulk(Wswz, sW, 128 * Nn);
  stage_wait_barrier();
  int wave = threadIdx.x >> 5, lane = threadIdx.x & 31;
  int tilesN = Nn >> 4;
  int mt = blockIdx.x * 8 + wave;
  if (mt < (M >> 4)) {
    v8f acc[8];
    for (int n = 0; n < tilesN; ++n) acc[n] = zero8();
    const f16* Ab = A + (size_t)mt * 16 * 128;
#pragma unroll
    for (int k0 = 0; k0 < 128; k0 += 32) {
      v16h a = load_a_frag(Ab + k0, 128);
      for (int n = 0; n < tilesN; ++n) {
        v16h b = load_b_swz(sW, (k0 >> 5) * tilesN + n);
        acc[n] = wmma16(a, b, acc[n]);
      }
    }
    int col = lane & 15, rbase = (lane >> 4) << 3;
    for (int n = 0; n < tilesN; ++n) {
#pragma unroll
      for (int v = 0; v < 8; ++v) {
        int row = mt * 16 + rbase + v, cc = n * 16 + col;
        float x = acc[n][v] + (bias ? bias[cc] : 0.f);
        if (ACT == 1) x = siluf(x);
        else if (ACT == 2) x = tanhf(x);
        else if (ACT == 3) x = fmaxf(x, 0.f);
        if (C32) C32[(size_t)row * ldc + cc] = x;
        if (C16) C16[(size_t)row * ldc + cc] = (f16)x;
      }
    }
  }
}

template <int ACT>
static void launch_gemm(const f16* A, const f16* Wswz, const float* bias, float* C32, f16* C16,
                        int ldc, int M, int Nn, hipStream_t s) {
  gemm_node_kernel<ACT><<<cdiv(M / 16, 8), 256, 0, s>>>(A, Wswz, bias, C32, C16, ldc, M, Nn);
}

// ---------------- fused EGNN layer ----------------
// LDS layout (bytes)
#define LB_SW1 0
#define LB_SW2 (LB_SW1 + 128 * 128 * 2)
#define LB_SA  (LB_SW2 + 128 * 128 * 2)
#define LB_SP  (LB_SA + 128 * 128 * 2)
#define LB_MS  (LB_SP + 128 * 128 * 2)
#define LB_D2  (LB_MS + 16 * 128 * 4)
#define LB_MK  (LB_D2 + 128 * 4)
#define LB_WC  (LB_MK + 128 * 4)
#define LB_CW2 (LB_WC + 128 * 4)
#define LB_EB2 (LB_CW2 + 128 * 4)
#define LB_CB1 (LB_EB2 + 128 * 4)
#define LB_NB2 (LB_CB1 + 128 * 4)
#define LB_REL (LB_NB2 + 128 * 4)
#define LB_SC  (LB_REL + 128 * 3 * 4)
#define LB_PD  (LB_SC + 128 * 4)
#define LB_CNT (LB_PD + 48 * 4)
#define LB_SJ  (LB_CNT + 16 * 4)
#define LB_TOT (LB_SJ + 128 * 4)

__global__ __launch_bounds__(256) void egnn_layer_kernel(
    const float* __restrict__ posIn, float* __restrict__ posOut,
    float* __restrict__ h32, f16* __restrict__ h16,
    const int* __restrict__ idx, const float* __restrict__ maskf,
    const float* __restrict__ Hi32, const float* __restrict__ Hj32,
    const float* __restrict__ Na32,
    const float* __restrict__ wc, const float* __restrict__ eb2,
    const float* __restrict__ cb1, const float* __restrict__ cw2,
    const float* __restrict__ nb2,
    const f16* __restrict__ wE2, const f16* __restrict__ wC1,
    const f16* __restrict__ wN1b, const f16* __restrict__ wN2) {
  extern __shared__ char smem[];
  f16*   sW1  = (f16*)(smem + LB_SW1);
  f16*   sW2  = (f16*)(smem + LB_SW2);
  f16*   sA   = (f16*)(smem + LB_SA);
  f16*   sP   = (f16*)(smem + LB_SP);
  float* sMs  = (float*)(smem + LB_MS);
  float* sD2  = (float*)(smem + LB_D2);
  float* sMk  = (float*)(smem + LB_MK);
  float* sWc  = (float*)(smem + LB_WC);
  float* sCw2 = (float*)(smem + LB_CW2);
  float* sEb2 = (float*)(smem + LB_EB2);
  float* sCb1 = (float*)(smem + LB_CB1);
  float* sNb2 = (float*)(smem + LB_NB2);
  float* sRel = (float*)(smem + LB_REL);
  float* sC   = (float*)(smem + LB_SC);
  float* sPd  = (float*)(smem + LB_PD);
  float* sCnt = (float*)(smem + LB_CNT);
  int*   sJ   = (int*)(smem + LB_SJ);

  const int t = threadIdx.x, lane = t & 31, wave = t >> 5;
  const int nb = blockIdx.x * 16;
  const int col = lane & 15, rbase = (lane >> 4) << 3;

  stage_wpair(wE2, sW1, wC1, sW2);
  for (int i = t; i < 16 * 128; i += 256) sMs[i] = 0.f;
  if (t < 128) { sWc[t] = wc[t]; sCw2[t] = cw2[t]; sEb2[t] = eb2[t]; sCb1[t] = cb1[t]; sNb2[t] = nb2[t]; }
  if (t < 48) sPd[t] = 0.f;
  if (t < 16) sCnt[t] = 0.f;
  stage_wpair_wait();

  for (int sub = 0; sub < 4; ++sub) {
    // stage per-edge scalars (128 edges = 4 nodes x 32 neighbors)
    if (t < 128) {
      int e = t;
      int i = nb + sub * 4 + (e >> 5);
      int ke = e & 31;
      int j = idx[i * KM + ke];
      float m = maskf[i * KM + ke];
      float rx = posIn[i * 3]     - posIn[j * 3];
      float ry = posIn[i * 3 + 1] - posIn[j * 3 + 1];
      float rz = posIn[i * 3 + 2] - posIn[j * 3 + 2];
      sJ[e] = j; sMk[e] = m;
      sRel[e * 3] = rx; sRel[e * 3 + 1] = ry; sRel[e * 3 + 2] = rz;
      sD2[e] = rx * rx + ry * ry + rz * rz;
      __builtin_prefetch(&Hj32[(size_t)j * 128], 0, 0);
    }
    __syncthreads();
    // m_pre = silu(HiPart[i] + HjPart[j] + d2*wc)  -> sA (f16)
    for (int id = t; id < 128 * 128; id += 256) {
      int e = id >> 7, c = id & 127;
      int i = nb + sub * 4 + (e >> 5);
      float v = Hi32[(size_t)i * 128 + c] + Hj32[(size_t)sJ[e] * 128 + c] + sD2[e] * sWc[c];
      sA[e * 128 + c] = (f16)siluf(v);
    }
    __syncthreads();

    // GEMM1: m = silu(sA @ ew2 + eb2) * mask   (WMMA f16)
    v8f acc[8];
#pragma unroll
    for (int n = 0; n < 8; ++n) acc[n] = zero8();
#pragma unroll
    for (int k0 = 0; k0 < 128; k0 += 32) {
      v16h a = load_a_frag(sA + wave * 16 * 128 + k0, 128);
#pragma unroll
      for (int n = 0; n < 8; ++n) {
        v16h b = load_b_swz(sW1, (k0 >> 5) * 8 + n);
        acc[n] = wmma16(a, b, acc[n]);
      }
    }
    __syncthreads();
    // epilogue: write masked m back to sA, and fold per-node message sums
    // directly from registers (half-wave shfl + LDS float atomics)
    {
      int nodeSl = (sub << 2) + (wave >> 1);  // this wave's 16 rows all belong to one node
#pragma unroll
      for (int n = 0; n < 8; ++n) {
        float psum = 0.f;
#pragma unroll
        for (int v = 0; v < 8; ++v) {
          int row = wave * 16 + rbase + v;
          int cc = n * 16 + col;
          float x = siluf(acc[n][v] + sEb2[cc]) * sMk[row];
          sA[row * 128 + cc] = (f16)x;
          psum += x;
        }
        psum += __shfl_xor(psum, 16, 32);
        if (lane < 16) atomicAdd(&sMs[nodeSl * 128 + n * 16 + col], psum);
      }
    }
    __syncthreads();

    // GEMM2: sP = silu(m @ cw1 + cb1)   (WMMA f16)
#pragma unroll
    for (int n = 0; n < 8; ++n) acc[n] = zero8();
#pragma unroll
    for (int k0 = 0; k0 < 128; k0 += 32) {
      v16h a = load_a_frag(sA + wave * 16 * 128 + k0, 128);
#pragma unroll
      for (int n = 0; n < 8; ++n) {
        v16h b = load_b_swz(sW2, (k0 >> 5) * 8 + n);
        acc[n] = wmma16(a, b, acc[n]);
      }
    }
#pragma unroll
    for (int n = 0; n < 8; ++n) {
#pragma unroll
      for (int v = 0; v < 8; ++v) {
        int row = wave * 16 + rbase + v;
        int cc = n * 16 + col;
        sP[row * 128 + cc] = (f16)siluf(acc[n][v] + sCb1[cc]);
      }
    }
    __syncthreads();

    // per-edge coord scalar: c_e = sP[e,:] . cw2   (vectorized LDS reads)
    if (t < 128) {
      float s = 0.f;
      const v8h* pr = (const v8h*)(sP + t * 128);
#pragma unroll
      for (int q = 0; q < 16; ++q) {
        v8h x = pr[q];
#pragma unroll
        for (int e = 0; e < 8; ++e) s += (float)x[e] * sCw2[q * 8 + e];
      }
      sC[t] = s;
    }
    __syncthreads();
    // pos delta + counts
    if (t < 128) {
      int nl = sub * 4 + (t >> 5);
      float wgt = sC[t] * sMk[t];
      atomicAdd(&sPd[nl * 3],     sRel[t * 3]     * wgt);
      atomicAdd(&sPd[nl * 3 + 1], sRel[t * 3 + 1] * wgt);
      atomicAdd(&sPd[nl * 3 + 2], sRel[t * 3 + 2] * wgt);
      atomicAdd(&sCnt[nl], sMk[t]);
    }
    __syncthreads();
  }

  // ---- node update ----
  stage_wpair(wN1b, sW1, wN2, sW2);
  for (int id = t; id < 16 * 128; id += 256) sA[id] = (f16)sMs[id];
  stage_wpair_wait();

  // GEMM3: T = silu(NodeHa + msum @ nw1_b)  (one 16x16 n-tile per wave)
  v8f a3 = zero8();
#pragma unroll
  for (int k0 = 0; k0 < 128; k0 += 32) {
    v16h a = load_a_frag(sA + k0, 128);
    v16h b = load_b_swz(sW1, (k0 >> 5) * 8 + wave);
    a3 = wmma16(a, b, a3);
  }
#pragma unroll
  for (int v = 0; v < 8; ++v) {
    int row = rbase + v, cc = wave * 16 + col;
    sP[row * 128 + cc] = (f16)siluf(a3[v] + Na32[(size_t)(nb + row) * 128 + cc]);
  }
  __syncthreads();

  // GEMM4: h += T @ nw2 + nb2
  v8f a4 = zero8();
#pragma unroll
  for (int k0 = 0; k0 < 128; k0 += 32) {
    v16h a = load_a_frag(sP + k0, 128);
    v16h b = load_b_swz(sW2, (k0 >> 5) * 8 + wave);
    a4 = wmma16(a, b, a4);
  }
#pragma unroll
  for (int v = 0; v < 8; ++v) {
    int row = rbase + v, cc = wave * 16 + col;
    size_t g = (size_t)(nb + row) * 128 + cc;
    float hn = h32[g] + a4[v] + sNb2[cc];
    h32[g] = hn;
    h16[g] = (f16)hn;
  }
  if (t < 48) {
    int nl = t / 3, d = t - nl * 3;
    float cnt = fmaxf(sCnt[nl], 1.f);
    posOut[(nb + nl) * 3 + d] = posIn[(nb + nl) * 3 + d] + sPd[t] / cnt;
  }
}

// ---------------- heads ----------------

__global__ void forces2_kernel(const f16* __restrict__ t1, const float* __restrict__ fw2,
                               const float* __restrict__ fb2, float* __restrict__ outF) {
  int n = blockIdx.x * blockDim.x + threadIdx.x;
  if (n >= NN) return;
  float a0 = fb2[0], a1 = fb2[1], a2 = fb2[2];
  for (int k = 0; k < 128; ++k) {
    float v = (float)t1[(size_t)n * 128 + k];
    a0 += v * fw2[k * 3]; a1 += v * fw2[k * 3 + 1]; a2 += v * fw2[k * 3 + 2];
  }
  outF[n * 3] = a0; outF[n * 3 + 1] = a1; outF[n * 3 + 2] = a2;
}

__global__ void uv_kernel(const float* __restrict__ h, const float* __restrict__ iw,
                          float* __restrict__ uv) {
  int n = blockIdx.x * blockDim.x + threadIdx.x;
  if (n >= NN) return;
  for (int tt = 0; tt < 3; ++tt) {
    const float* wt = iw + tt * 258;
    float u = 0.f, v = 0.f;
    for (int c = 0; c < 128; ++c) {
      float hv = h[(size_t)n * 128 + c];
      u += hv * wt[c];
      v += hv * wt[128 + c];
    }
    uv[n * 6 + tt * 2] = u;
    uv[n * 6 + tt * 2 + 1] = v;
  }
}

__global__ void scores_kernel(const float* __restrict__ pos, const int* __restrict__ idx,
                              const float* __restrict__ mskf, const float* __restrict__ uv,
                              const float* __restrict__ iw, const float* __restrict__ ib,
                              float* __restrict__ s0, float* __restrict__ s1,
                              float* __restrict__ s2) {
  int id = blockIdx.x * blockDim.x + threadIdx.x;
  if (id >= NN * KI) return;
  int n = id / KI, k = id - n * KI;
  int j = idx[n * KM + k];
  float m = mskf[n * KM + k];
  float dx = pos[n * 3] - pos[j * 3];
  float dy = pos[n * 3 + 1] - pos[j * 3 + 1];
  float dz = pos[n * 3 + 2] - pos[j * 3 + 2];
  float dist = sqrtf(dx * dx + dy * dy + dz * dz + 1e-12f);
  float* outs[3] = {s0, s1, s2};
#pragma unroll
  for (int tt = 0; tt < 3; ++tt) {
    float val = uv[n * 6 + tt * 2] + uv[j * 6 + tt * 2 + 1] +
                dist * iw[tt * 258 + 256] + dist * 0.1f * iw[tt * 258 + 257] + ib[tt];
    outs[tt][id] = m / (1.f + __expf(-val));
  }
}

__global__ void steric_kernel(const float* __restrict__ h, const float* __restrict__ sw,
                              const float* __restrict__ sb, float* __restrict__ o) {
  int n = blockIdx.x * blockDim.x + threadIdx.x;
  if (n >= NN) return;
  float a = sb[0];
  for (int k = 0; k < 128; ++k) a += h[(size_t)n * 128 + k] * sw[k];
  o[n] = a;
}

// ---------------- host ----------------

extern "C" void kernel_launch(void* const* d_in, const int* in_sizes, int n_in,
                              void* d_out, int out_size, void* d_ws, size_t ws_size,
                              hipStream_t stream) {
  (void)in_sizes; (void)n_in; (void)out_size; (void)ws_size;

  const float* H_IN  = (const float*)d_in[0];
  const float* P_IN  = (const float*)d_in[1];
  const int*   BATCH = (const int*)d_in[2];
  const float* EW1 = (const float*)d_in[3];
  const float* EB1 = (const float*)d_in[4];
  const float* EW2 = (const float*)d_in[5];
  const float* EB2 = (const float*)d_in[6];
  const float* CW1 = (const float*)d_in[7];
  const float* CB1 = (const float*)d_in[8];
  const float* CW2 = (const float*)d_in[9];
  const float* NW1 = (const float*)d_in[10];
  const float* NB1 = (const float*)d_in[11];
  const float* NW2 = (const float*)d_in[12];
  const float* NB2 = (const float*)d_in[13];
  const float* FW1 = (const float*)d_in[14];
  const float* FB1 = (const float*)d_in[15];
  const float* FW2 = (const float*)d_in[16];
  const float* FB2 = (const float*)d_in[17];
  const float* IW  = (const float*)d_in[18];
  const float* IB  = (const float*)d_in[19];
  const float* CH1 = (const float*)d_in[20];
  const float* CHB1 = (const float*)d_in[21];
  const float* CH2 = (const float*)d_in[22];
  const float* CHB2 = (const float*)d_in[23];
  const float* SWp = (const float*)d_in[24];
  const float* SBp = (const float*)d_in[25];

  float* out = (float*)d_out;
  float* h32    = out;                     // [N,H] working h == output h
  float* outPos = out + (size_t)NN * HH;
  float* outF   = outPos + NN * 3;
  float* outS0  = outF + NN * 3;
  float* outS1  = outS0 + NN * KI;
  float* outS2  = outS1 + NN * KI;
  float* outC   = outS2 + NN * KI;
  float* outSt  = outC + NN * 64;

  char* w = (char*)d_ws;
  size_t off = 0;
  auto carve = [&](size_t bytes) -> char* {
    char* p = w + off;
    off = (off + bytes + 255) & ~(size_t)255;
    return p;
  };

  f16*   h16  = (f16*)carve((size_t)NN * HH * 2);
  float* posA = (float*)carve(NN * 3 * 4);
  float* posB = (float*)carve(NN * 3 * 4);
  int*   idx  = (int*)carve((size_t)NN * KM * 4);
  float* mskf = (float*)carve((size_t)NN * KM * 4);
  float* Hi32 = (float*)carve((size_t)NN * HH * 4);
  float* Hj32 = (float*)carve((size_t)NN * HH * 4);
  float* Na32 = (float*)carve((size_t)NN * HH * 4);
  f16* wE1a = (f16*)carve(HH * HH * 2);
  f16* wE1b = (f16*)carve(HH * HH * 2);
  f16* wE2s = (f16*)carve(HH * HH * 2);
  f16* wC1s = (f16*)carve(HH * HH * 2);
  f16* wN1a = (f16*)carve(HH * HH * 2);
  f16* wN1b = (f16*)carve(HH * HH * 2);
  f16* wN2s = (f16*)carve(HH * HH * 2);
  f16* wF1s  = (f16*)carve(HH * HH * 2);
  f16* wCH1s = (f16*)carve(HH * HH * 2);
  f16* wCH2s = (f16*)carve(HH * 64 * 2);
  // reuse dead buffers for head intermediates
  f16*   t1_16  = (f16*)Hi32;
  f16*   cf1_16 = (f16*)Hj32;
  float* sUV    = Na32;  // [N,6]

  (void)hipFuncSetAttribute((const void*)egnn_layer_kernel,
                            hipFuncAttributeMaxDynamicSharedMemorySize, LB_TOT);

  // init working state
  (void)hipMemcpyAsync(h32, H_IN, (size_t)NN * HH * 4, hipMemcpyDeviceToDevice, stream);
  (void)hipMemcpyAsync(posA, P_IN, NN * 3 * 4, hipMemcpyDeviceToDevice, stream);
  cvt_kernel<<<cdiv(NN * HH, 256), 256, 0, stream>>>(H_IN, h16, NN * HH);

  float* pin = posA;
  float* pout = posB;
  const float cut[6] = {3.f, 3.f, 6.f, 6.f, 10.f, 10.f};

  for (int l = 0; l < 6; ++l) {
    if ((l & 1) == 0)
      neighbors_kernel<<<NN / 4, 128, 0, stream>>>(pin, BATCH, cut[l] * cut[l], KM, idx, mskf);

    const float* ew1l = EW1 + (size_t)l * 257 * 128;
    cvt_swz_kernel<<<64, 256, 0, stream>>>(ew1l, wE1a, 128, 128);
    cvt_swz_kernel<<<64, 256, 0, stream>>>(ew1l + 16384, wE1b, 128, 128);
    cvt_swz_kernel<<<64, 256, 0, stream>>>(EW2 + (size_t)l * 16384, wE2s, 128, 128);
    cvt_swz_kernel<<<64, 256, 0, stream>>>(CW1 + (size_t)l * 16384, wC1s, 128, 128);
    cvt_swz_kernel<<<64, 256, 0, stream>>>(NW1 + (size_t)l * 256 * 128, wN1a, 128, 128);
    cvt_swz_kernel<<<64, 256, 0, stream>>>(NW1 + (size_t)l * 256 * 128 + 16384, wN1b, 128, 128);
    cvt_swz_kernel<<<64, 256, 0, stream>>>(NW2 + (size_t)l * 16384, wN2s, 128, 128);

    // per-node precomputed halves of the edge/node MLP first layers
    launch_gemm<0>(h16, wE1a, EB1 + l * 128, Hi32, (f16*)nullptr, 128, NN, 128, stream);
    launch_gemm<0>(h16, wE1b, (const float*)nullptr, Hj32, (f16*)nullptr, 128, NN, 128, stream);
    launch_gemm<0>(h16, wN1a, NB1 + l * 128, Na32, (f16*)nullptr, 128, NN, 128, stream);

    egnn_layer_kernel<<<NN / 16, 256, LB_TOT, stream>>>(
        pin, pout, h32, h16, idx, mskf, Hi32, Hj32, Na32,
        ew1l + 256 * 128, EB2 + l * 128, CB1 + l * 128, CW2 + l * 128, NB2 + l * 128,
        wE2s, wC1s, wN1b, wN2s);

    float* tmp = pin; pin = pout; pout = tmp;
  }

  // heads
  cvt_swz_kernel<<<64, 256, 0, stream>>>(FW1, wF1s, 128, 128);
  cvt_swz_kernel<<<64, 256, 0, stream>>>(CH1, wCH1s, 128, 128);
  cvt_swz_kernel<<<32, 256, 0, stream>>>(CH2, wCH2s, 128, 64);

  launch_gemm<2>(h16, wF1s, FB1, (float*)nullptr, t1_16, 128, NN, 128, stream);
  forces2_kernel<<<cdiv(NN, 256), 256, 0, stream>>>(t1_16, FW2, FB2, outF);

  launch_gemm<3>(h16, wCH1s, CHB1, (float*)nullptr, cf1_16, 128, NN, 128, stream);
  launch_gemm<0>(cf1_16, wCH2s, CHB2, outC, (f16*)nullptr, 64, NN, 64, stream);

  neighbors_kernel<<<NN / 4, 128, 0, stream>>>(pin, BATCH, 100.f, KI, idx, mskf);
  uv_kernel<<<cdiv(NN, 256), 256, 0, stream>>>(h32, IW, sUV);
  scores_kernel<<<cdiv(NN * KI, 256), 256, 0, stream>>>(pin, idx, mskf, sUV, IW, IB,
                                                        outS0, outS1, outS2);
  steric_kernel<<<cdiv(NN, 256), 256, 0, stream>>>(h32, SWp, SBp, outSt);

  (void)hipMemcpyAsync(outPos, pin, NN * 3 * 4, hipMemcpyDeviceToDevice, stream);
}
